// Net2_68092411510900
// MI455X (gfx1250) — compile-verified, matrix operand order checked by reading, not run on hardware
//
#include <hip/hip_runtime.h>

#define Dz 32
#define Hy 512
#define Wx 512
#define GRID_VOX (Dz * Hy * Wx)

typedef __attribute__((ext_vector_type(16))) __bf16 v16bf;
typedef __attribute__((ext_vector_type(8)))  float  v8f;

union Frag32B {
    uint4 u[2];
    v16bf v;
};

__device__ __forceinline__ unsigned short f2bf(float f) {
    unsigned int u = __float_as_uint(f);
    unsigned int r = u + 0x7FFFu + ((u >> 16) & 1u);   // round-to-nearest-even
    return (unsigned short)(r >> 16);
}

// ---------------- grid / neighbor construction ----------------

__global__ void k_init_grid(int* grid) {
    int i = blockIdx.x * blockDim.x + threadIdx.x;
    if (i < GRID_VOX) grid[i] = -1;
}

__global__ void k_zero256(unsigned short* z) {
    if (threadIdx.x < 128) z[threadIdx.x] = 0;
}

__global__ void k_scatter(const int* coors, int* grid, int N) {
    int i = blockIdx.x * blockDim.x + threadIdx.x;
    if (i >= N) return;
    int z = coors[i * 4 + 1], y = coors[i * 4 + 2], x = coors[i * 4 + 3];
    grid[(z * Hy + y) * Wx + x] = i;
}

__global__ void k_neighbors(const int* coors, const int* grid, int* nbr, int N, int NP) {
    int i = blockIdx.x * blockDim.x + threadIdx.x;
    if (i >= N) return;
    int z = coors[i * 4 + 1], y = coors[i * 4 + 2], x = coors[i * 4 + 3];
    for (int k = 0; k < 27; ++k) {
        int dz = k / 9 - 1, dy = (k / 3) % 3 - 1, dx = k % 3 - 1;
        int nz = z + dz, ny = y + dy, nx = x + dx;
        int v = -1;
        if (nz >= 0 && nz < Dz && ny >= 0 && ny < Hy && nx >= 0 && nx < Wx)
            v = grid[(nz * Hy + ny) * Wx + nx];
        nbr[k * NP + i] = v;
    }
}

// ---------------- layer 1: Cin=3 -> 128, f32 FMA, bf16 out ----------------

__global__ __launch_bounds__(128) void k_layer1(const float* __restrict__ feat,
                                                const float* __restrict__ w1,
                                                const int* __restrict__ nbr,
                                                unsigned short* __restrict__ h,
                                                int N, int NP) {
    __shared__ float w1s[27 * 3 * 128];   // 41472 B
    for (int idx = threadIdx.x; idx < 27 * 3 * 128; idx += 128) w1s[idx] = w1[idx];
    __syncthreads();
    int c = threadIdx.x;
    int base = blockIdx.x * 32;
    for (int v = 0; v < 32; ++v) {
        int i = base + v;
        if (i >= N) break;
        float acc = 0.f;
        #pragma unroll 1
        for (int k = 0; k < 27; ++k) {
            int nidx = nbr[k * NP + i];
            if (nidx >= 0) {
                const float* f = feat + (size_t)nidx * 3;
                float f0 = f[0], f1 = f[1], f2 = f[2];
                const float* ws = w1s + k * 384 + c;
                acc = fmaf(f0, ws[0],   acc);
                acc = fmaf(f1, ws[128], acc);
                acc = fmaf(f2, ws[256], acc);
            }
        }
        h[(size_t)i * 128 + c] = f2bf(acc);
    }
}

// ---------------- pack w2 into WMMA B-fragment layout (bf16) ----------------
// packed index p = (((k*4 + kk)*8 + nt)*32 + lane)*16 + j
// lane<16 : col n = nt*16+lane      , K = kk*32 + (j&7) + (j>>3)*16
// lane>=16: col n = nt*16+(lane-16) , K = kk*32 + 8 + (j&7) + (j>>3)*16

__global__ void k_pack_w2(const float* __restrict__ w2, unsigned short* __restrict__ w2p) {
    int tid = blockIdx.x * blockDim.x + threadIdx.x;
    if (tid >= 27 * 128 * 128) return;
    int j    = tid & 15;
    int lane = (tid >> 4) & 31;
    int nt   = (tid >> 9) & 7;
    int kk   = (tid >> 12) & 3;
    int k    = tid >> 14;
    int Klocal = ((lane >> 4) & 1) * 8 + (j & 7) + ((j >> 3) << 4);
    int K = kk * 32 + Klocal;
    int n = nt * 16 + (lane & 15);
    w2p[tid] = f2bf(w2[((size_t)k * 128 + K) * 128 + n]);
}

// ---------------- layer 2: 27x gathered 128x128 bf16 GEMM via WMMA ----------------
// Double-buffered LDS (2 x 128 rows x 256B = 64KB), gather via
// GLOBAL_LOAD_ASYNC_TO_LDS_B128 (ASYNCcnt) overlapped with WMMA compute.
// 16B segments XOR-swizzled within each 256B row: slot = seg ^ (row & 15),
// so A-fragment ds_load_b128 across 16 rows hits each bank-quad once.

#define BUFSZ (128 * 256)

#define GASYNC(LOFF, GA) \
    asm volatile("global_load_async_to_lds_b128 %0, %1, off" :: "v"(LOFF), "v"(GA) : "memory")

__global__ __launch_bounds__(256) void k_layer2(const unsigned short* __restrict__ h,
                                                const unsigned short* __restrict__ w2p,
                                                const int* __restrict__ nbr,
                                                const unsigned short* __restrict__ zbuf,
                                                float* __restrict__ out,
                                                int N, int NP) {
    __shared__ __align__(16) unsigned char smem[2 * BUFSZ];   // 65536 B
    const int lane = threadIdx.x & 31;
    const int wv   = threadIdx.x >> 5;      // wave id == N-tile (0..7)
    const int arow = lane & 15;
    const int ahi  = (lane >> 4) & 1;
    const int base = blockIdx.x * 128;
    const int r    = threadIdx.x >> 1;      // gather row (0..127)
    const int half = threadIdx.x & 1;       // gather half-row
    const unsigned smembase = (unsigned)(uintptr_t)(void*)smem;  // LDS byte offset

    v8f C[8];
    #pragma unroll
    for (int mt = 0; mt < 8; ++mt) C[mt] = (v8f){0.f, 0.f, 0.f, 0.f, 0.f, 0.f, 0.f, 0.f};

    auto issue_gather = [&](int kidx, unsigned bufofs) {
        int gi = base + r;
        int nidx = (gi < N) ? nbr[kidx * NP + gi] : -1;
        unsigned long long gbase = (nidx >= 0)
            ? (unsigned long long)(uintptr_t)((const unsigned char*)h + (size_t)nidx * 256)
            : (unsigned long long)(uintptr_t)zbuf;
        unsigned rowbase = smembase + bufofs + (unsigned)(r * 256);
        int r15 = r & 15;
        #pragma unroll
        for (int qi = 0; qi < 8; ++qi) {
            int q = half * 8 + qi;                       // 16B segment index (0..15)
            unsigned loff = rowbase + (unsigned)(((q ^ r15) & 15) * 16);
            unsigned long long ga = gbase + (unsigned)(q * 16);
            GASYNC(loff, ga);
        }
    };

    issue_gather(0, 0u);   // prologue: k=0 -> buffer 0

    #pragma unroll 1
    for (int k = 0; k < 27; ++k) {
        unsigned cur = (unsigned)(k & 1) * BUFSZ;
        asm volatile("s_wait_asynccnt 0x0" ::: "memory");  // my gather landed
        __syncthreads();                                    // everyone's landed
        if (k + 1 < 27) issue_gather(k + 1, (unsigned)((k + 1) & 1) * BUFSZ);

        #pragma unroll
        for (int kk = 0; kk < 4; ++kk) {
            Frag32B b;
            const uint4* bp = (const uint4*)(w2p + ((((size_t)k * 4 + kk) * 8 + wv) * 32 + lane) * 16);
            b.u[0] = bp[0];
            b.u[1] = bp[1];
            int seg0 = kk * 4 + ahi;                    // 16B segment of first A half
            #pragma unroll
            for (int mt = 0; mt < 8; ++mt) {
                int row = mt * 16 + arow;
                const unsigned char* rp = smem + cur + row * 256;
                Frag32B a;
                a.u[0] = *(const uint4*)(rp + (((seg0    ) ^ arow) & 15) * 16);
                a.u[1] = *(const uint4*)(rp + (((seg0 + 2) ^ arow) & 15) * 16);
                C[mt] = __builtin_amdgcn_wmma_f32_16x16x32_bf16(
                    false, a.v, false, b.v, (short)0, C[mt], false, false);
            }
        }
        __syncthreads();   // done reading cur before it is refilled at k+2
    }

    // C/D layout: VGPR j -> row (mt*16 + 8*ahi + j), col = wv*16 + (lane&15)
    int col = wv * 16 + (lane & 15);
    int rofs = 8 * ahi;
    #pragma unroll
    for (int mt = 0; mt < 8; ++mt) {
        #pragma unroll
        for (int j = 0; j < 8; ++j) {
            int row = base + mt * 16 + rofs + j;
            if (row < N) out[(size_t)row * 128 + col] = C[mt][j];
        }
    }
}

// ---------------- host launcher ----------------

extern "C" void kernel_launch(void* const* d_in, const int* in_sizes, int n_in,
                              void* d_out, int out_size, void* d_ws, size_t ws_size,
                              hipStream_t stream) {
    const float* feat  = (const float*)d_in[0];
    const int*   coors = (const int*)d_in[1];
    const float* w1    = (const float*)d_in[2];
    const float* w2    = (const float*)d_in[3];
    float*       out   = (float*)d_out;

    const int N  = in_sizes[0] / 3;
    const int NP = ((N + 127) / 128) * 128;

    char* ws = (char*)d_ws;
    size_t off = 0;
    int* grid = (int*)(ws + off);                       off += (size_t)GRID_VOX * 4;
    int* nbr  = (int*)(ws + off);                       off += ((size_t)27 * NP * 4 + 255) & ~(size_t)255;
    unsigned short* hbuf = (unsigned short*)(ws + off); off += ((size_t)NP * 128 * 2 + 255) & ~(size_t)255;
    unsigned short* w2p  = (unsigned short*)(ws + off); off += ((size_t)27 * 128 * 128 * 2 + 255) & ~(size_t)255;
    unsigned short* zbuf = (unsigned short*)(ws + off); // 256 B of zeros

    k_init_grid<<<(GRID_VOX + 255) / 256, 256, 0, stream>>>(grid);
    k_zero256<<<1, 128, 0, stream>>>(zbuf);
    k_scatter<<<(N + 255) / 256, 256, 0, stream>>>(coors, grid, N);
    k_neighbors<<<(N + 255) / 256, 256, 0, stream>>>(coors, grid, nbr, N, NP);
    k_pack_w2<<<(27 * 128 * 128 + 255) / 256, 256, 0, stream>>>(w2, w2p);
    k_layer1<<<(N + 31) / 32, 128, 0, stream>>>(feat, w1, nbr, hbuf, N, NP);
    k_layer2<<<NP / 128, 256, 0, stream>>>(hbuf, w2p, nbr, zbuf, out, N, NP);
}
